// SimpleCNN_62002147885323
// MI455X (gfx1250) — compile-verified, matrix operand order checked by reading, not run on hardware
//
#include <hip/hip_runtime.h>
#include <hip/hip_bf16.h>
#include <math.h>

// ---------------------------------------------------------------------------
// CDNA5 (gfx1250) bf16 WMMA implementation of FE(MLP) + 2x BiLSTM + FC.
// wave32 only. Matrix core: v_wmma_f32_16x16x32_bf16.
// Async staging: global_load_async_to_lds_b64 + s_wait_asynccnt (ISA 15.18.3).
// ---------------------------------------------------------------------------

#define USE_ASYNC 1

typedef __attribute__((ext_vector_type(16))) __bf16 v16bf;
typedef __attribute__((ext_vector_type(8)))  float  v8f;

union ABFrag { v16bf v; uint4 q[2]; };

// WMMA bf16 fragment = two contiguous 16B runs per lane:
//   K in [base, base+8) and [base+16, base+24), base = k0 + 8*hi.
__device__ __forceinline__ void load_frag_global(ABFrag& f, const __bf16* base) {
    const uint4* p = (const uint4*)base;   // base must be 16B aligned
    f.q[0] = p[0];
    f.q[1] = p[2];                          // +16 bf16 elements = +32B
}

__device__ __forceinline__ float sigf(float x) {
    return 1.0f / (1.0f + __expf(-x));
}

// Opaque zero: defeats LICM so loads addressed with (ptr + opaque_zero())
// stay inside the loop (streamed from L2) instead of being hoisted into
// resident VGPRs and spilled to scratch.
__device__ __forceinline__ int opaque_zero() {
    int z;
    asm volatile("s_mov_b32 %0, 0" : "=s"(z));
    return z;
}

#if USE_ASYNC
__device__ __forceinline__ void async_ld_lds_b64(void* lds_ptr, const void* gptr) {
    uint32_t lds_off = (uint32_t)(uintptr_t)lds_ptr;  // LDS aperture: addr[31:0]
    asm volatile("global_load_async_to_lds_b64 %0, %1, off"
                 :: "v"(lds_off), "v"((uint64_t)(uintptr_t)gptr)
                 : "memory");
}
#endif

// ---------------------------------------------------------------------------
// f32 -> bf16 convert with zero padding of rows/cols.
// ---------------------------------------------------------------------------
__global__ void padcvt_bf16(__bf16* __restrict__ dst, const float* __restrict__ src,
                            int rd, int cd, int rs, int cs) {
    int i = blockIdx.x * 256 + threadIdx.x;
    if (i >= rd * cd) return;
    int r = i / cd, c = i % cd;
    float v = (r < rs && c < cs) ? src[(size_t)r * cs + c] : 0.0f;
    dst[i] = (__bf16)v;
}

__global__ void add_bias_pair(float* __restrict__ dst, const float* __restrict__ a,
                              const float* __restrict__ b, int n) {
    int i = blockIdx.x * 256 + threadIdx.x;
    if (i < n) dst[i] = a[i] + b[i];
}

// ---------------------------------------------------------------------------
// bf16 WMMA GEMM:  C[M,N] = act(A[M,K] * B[N,K]^T + bias[N])
// Block: 256 threads = 8 waves; block tile 32(M) x 256(N);
// wave tile 16(M) x 64(N). A staged through LDS in 32-wide K chunks,
// double-buffered with async global->LDS copies; B streamed from L2.
// ---------------------------------------------------------------------------
template <bool RELU, bool BF16OUT>
__global__ __launch_bounds__(256) void gemm_bf16_wmma(
    const __bf16* __restrict__ A, int lda,
    const __bf16* __restrict__ B, int ldb,
    const float* __restrict__ bias,
    void* __restrict__ C, int ldc,
    int M, int N, int K, int Nvalid) {
    const int tid  = threadIdx.x;
    const int lane = tid & 31;
    const int w    = tid >> 5;
    const int wy   = w >> 2, wx = w & 3;
    const int r    = lane & 15, hi = lane >> 4;
    const int m0b  = blockIdx.y * 32;           // block M base
    const int nb   = blockIdx.x * 256 + wx * 64;

    __shared__ __align__(16) __bf16 At[2][32][32];   // 2 x (32 rows x 32 K) bf16

    // Staging role: each of the 256 threads copies 8B (4 bf16) per chunk.
    const int srow = tid >> 3;        // 0..31
    const int sseg = tid & 7;         // 0..7 -> K offset sseg*4

    auto stage = [&](int buf, int k0) {
        const __bf16* g = A + (size_t)(m0b + srow) * lda + k0 + sseg * 4;
#if USE_ASYNC
        async_ld_lds_b64(&At[buf][srow][sseg * 4], g);
#else
        *(uint2*)&At[buf][srow][sseg * 4] = *(const uint2*)g;
#endif
    };

    v8f acc[4];
#pragma unroll
    for (int nt = 0; nt < 4; ++nt)
#pragma unroll
        for (int v = 0; v < 8; ++v) acc[nt][v] = 0.0f;

    const int nc = K >> 5;            // K chunks of 32 (all K are multiples of 32)
    stage(0, 0);
    for (int ci = 0; ci < nc; ++ci) {
        if (ci + 1 < nc) stage((ci + 1) & 1, (ci + 1) * 32);
#if USE_ASYNC
        if (ci + 1 < nc) asm volatile("s_wait_asynccnt 0x1" ::: "memory");
        else             asm volatile("s_wait_asynccnt 0x0" ::: "memory");
#endif
        __syncthreads();

        // A fragment from LDS (row = wy*16 + r): two b128 reads.
        ABFrag a;
        const char* hrow = (const char*)&At[ci & 1][wy * 16 + r][0];
        a.q[0] = *(const uint4*)(hrow + 16 * hi);
        a.q[1] = *(const uint4*)(hrow + 32 + 16 * hi);

        const int k0 = ci * 32;
#pragma unroll
        for (int nt = 0; nt < 4; ++nt) {
            int n0 = nb + nt * 16;
            if (n0 >= N) continue;    // wave-uniform
            ABFrag b;
            load_frag_global(b, B + (size_t)(n0 + r) * ldb + k0 + 8 * hi);
            acc[nt] = __builtin_amdgcn_wmma_f32_16x16x32_bf16(
                false, a.v, false, b.v, (short)0, acc[nt], false, false);
        }
        __syncthreads();
    }

#pragma unroll
    for (int nt = 0; nt < 4; ++nt) {
        int n = nb + nt * 16 + r;
        if (n >= Nvalid) continue;
        float bv = bias[n];
#pragma unroll
        for (int v = 0; v < 8; ++v) {
            int m = m0b + wy * 16 + v + 8 * hi;
            float x = acc[nt][v] + bv;
            if (RELU) x = fmaxf(x, 0.0f);
            if (BF16OUT)
                ((__bf16*)C)[(size_t)m * ldc + n] = (__bf16)x;
            else
                ((float*)C)[(size_t)m * ldc + n] = x;
        }
    }
}

// ---------------------------------------------------------------------------
// Persistent bidirectional LSTM recurrence. grid = 2 (dir), block = 512.
// Wave w (of 16) owns hidden cols [16w,16w+16) and the 4 gate tiles
// {g*256+16w}. W_hh streams from the 192MB L2 every step (512KB/dir, hot);
// the opaque_zero() trick keeps those loads in-loop (no hoist -> no spill).
// h[16,256] ping-pongs through LDS as bf16 (padded row 264 -> 16B-aligned,
// bank-staggered). xg[t+1] is prefetched during step t (global_prefetch_b8).
// Per step: 16 ds_load_b128 + 64 global b128 + 32 b32 + 32 WMMA.
// xg: x@W_ih^T + (b_ih+b_hh), layout [(b*T+t), 1024], gates i,f,g,o.
// out: bf16 concat [(b*T+t), 512]; fwd -> cols 0..255, bwd -> 256..511.
// ---------------------------------------------------------------------------
__global__ __launch_bounds__(512) void lstm_recurrence(
    const float* __restrict__ xg_f, const float* __restrict__ xg_b,
    const __bf16* __restrict__ whh_f, const __bf16* __restrict__ whh_b,
    __bf16* __restrict__ out, int T) {
    const int dir = blockIdx.x;
    const float* xg   = dir ? xg_b : xg_f;
    const __bf16* whh = dir ? whh_b : whh_f;
    const int colofs  = dir ? 256 : 0;

    const int lane = threadIdx.x & 31;
    const int w    = threadIdx.x >> 5;
    const int r    = lane & 15, hi = lane >> 4;

    __shared__ __align__(16) __bf16 hbuf[2][16][264];   // 264*2B = 528B = 33x16B
    for (int i = threadIdx.x; i < 16 * 264; i += 512)
        hbuf[0][i / 264][i % 264] = (__bf16)0.0f;

    v8f cst;
#pragma unroll
    for (int v = 0; v < 8; ++v) cst[v] = 0.0f;

    __syncthreads();

    int cur = 0;
    for (int s = 0; s < T; ++s) {
        const int t = dir ? (T - 1 - s) : s;

        // Loop-variant (opaque) weight base: forces per-step L2 streaming.
        const __bf16* whh_t = whh + opaque_zero();

        // Prefetch next step's x-gate tile while this step computes.
        if (s + 1 < T) {
            const int tn = dir ? (T - 2 - s) : (s + 1);
#pragma unroll
            for (int j = 0; j < 2; ++j) {
                int pg = hi + 2 * j;   // wave covers 16 rows x 4 gate segments
                __builtin_prefetch(
                    &xg[((size_t)r * T + tn) * 1024 + pg * 256 + 16 * w], 0, 3);
            }
        }

        // A fragments (h_prev: M=batch, K=hidden) from LDS: 2 b128 per K-tile.
        ABFrag af[8];
        const char* hrow = (const char*)&hbuf[cur][r][0];
#pragma unroll
        for (int kt = 0; kt < 8; ++kt) {
            af[kt].q[0] = *(const uint4*)(hrow + kt * 64 + 16 * hi);
            af[kt].q[1] = *(const uint4*)(hrow + kt * 64 + 32 + 16 * hi);
        }

        // acc init = x-gate contribution, then 8 streamed-weight WMMAs/gate.
        v8f acc[4];
#pragma unroll
        for (int g = 0; g < 4; ++g) {
#pragma unroll
            for (int v = 0; v < 8; ++v) {
                int b = v + 8 * hi;
                acc[g][v] = xg[((size_t)b * T + t) * 1024 + g * 256 + 16 * w + r];
            }
            const __bf16* wrow = whh_t + (size_t)(g * 256 + 16 * w + r) * 256;
#pragma unroll
            for (int kt = 0; kt < 8; ++kt) {
                ABFrag b;
                load_frag_global(b, wrow + kt * 32 + 8 * hi);
                acc[g] = __builtin_amdgcn_wmma_f32_16x16x32_bf16(
                    false, af[kt].v, false, b.v, (short)0, acc[g], false, false);
            }
        }

        // LSTM cell (lane = hidden col 16w+r, VGPR v = batch row).
        const int nxt = cur ^ 1;
#pragma unroll
        for (int v = 0; v < 8; ++v) {
            int b = v + 8 * hi;
            float iv = sigf(acc[0][v]);
            float fv = sigf(acc[1][v]);
            float gv = tanhf(acc[2][v]);
            float ov = sigf(acc[3][v]);
            cst[v] = fv * cst[v] + iv * gv;
            float hv = ov * tanhf(cst[v]);
            __bf16 hb = (__bf16)hv;
            hbuf[nxt][b][16 * w + r] = hb;
            out[((size_t)b * T + t) * 512 + colofs + 16 * w + r] = hb;
        }
        __syncthreads();
        cur = nxt;
    }
}

// ---------------------------------------------------------------------------
// Host-side orchestration.
// ---------------------------------------------------------------------------
static inline size_t alignup(size_t x) { return (x + 255) & ~(size_t)255; }

extern "C" void kernel_launch(void* const* d_in, const int* in_sizes, int n_in,
                              void* d_out, int out_size, void* d_ws, size_t ws_size,
                              hipStream_t stream) {
    (void)in_sizes; (void)n_in; (void)out_size; (void)ws_size;
    const float* x     = (const float*)d_in[0];
    const float* fe_w1 = (const float*)d_in[1];
    const float* fe_b1 = (const float*)d_in[2];
    const float* fe_w2 = (const float*)d_in[3];
    const float* fe_b2 = (const float*)d_in[4];
    const float* wih[4] = {(const float*)d_in[5],  (const float*)d_in[9],
                           (const float*)d_in[13], (const float*)d_in[17]};
    const float* whh[4] = {(const float*)d_in[6],  (const float*)d_in[10],
                           (const float*)d_in[14], (const float*)d_in[18]};
    const float* bih[4] = {(const float*)d_in[7],  (const float*)d_in[11],
                           (const float*)d_in[15], (const float*)d_in[19]};
    const float* bhh[4] = {(const float*)d_in[8],  (const float*)d_in[12],
                           (const float*)d_in[16], (const float*)d_in[20]};
    const float* fc_w = (const float*)d_in[21];
    const float* fc_b = (const float*)d_in[22];

    const int Bsz = 16, T = 1000, M = Bsz * T;  // 16000 rows

    char* ws = (char*)d_ws;
    size_t off = 0;
    auto alloc = [&](size_t bytes) { void* p = ws + off; off += alignup(bytes); return p; };

    __bf16* xpad  = (__bf16*)alloc((size_t)M * 96 * 2);
    __bf16* w1p   = (__bf16*)alloc((size_t)1024 * 96 * 2);
    __bf16* w2b   = (__bf16*)alloc((size_t)2560 * 1024 * 2);
    __bf16* wih0f = (__bf16*)alloc((size_t)1024 * 2560 * 2);
    __bf16* wih0b = (__bf16*)alloc((size_t)1024 * 2560 * 2);
    __bf16* wih1f = (__bf16*)alloc((size_t)1024 * 512 * 2);
    __bf16* wih1b = (__bf16*)alloc((size_t)1024 * 512 * 2);
    __bf16* whhc[4];
    for (int i = 0; i < 4; ++i) whhc[i] = (__bf16*)alloc((size_t)1024 * 256 * 2);
    __bf16* fcwp  = (__bf16*)alloc((size_t)32 * 512 * 2);
    float*  bcomb[4];
    for (int i = 0; i < 4; ++i) bcomb[i] = (float*)alloc(1024 * 4);
    __bf16* Y1  = (__bf16*)alloc((size_t)M * 1024 * 2);   // FE1 out; reused as concat1
    __bf16* Y2  = (__bf16*)alloc((size_t)M * 2560 * 2);   // FE2 out; reused as concat0
    float*  xgf = (float*)alloc((size_t)M * 1024 * 4);    // x-gates fwd (l0 then l1)
    float*  xgb = (float*)alloc((size_t)M * 1024 * 4);    // x-gates bwd (l0 then l1)
    __bf16* concat0 = Y2;
    __bf16* concat1 = Y1;

    auto cvt = [&](__bf16* dst, const float* src, int rd, int cd, int rs, int cs) {
        int n = rd * cd;
        padcvt_bf16<<<(n + 255) / 256, 256, 0, stream>>>(dst, src, rd, cd, rs, cs);
    };

    // --- weight / input conversion to bf16 (pad K:80->96, fc rows:29->32) ---
    cvt(xpad,  x,      M,    96,  M,    80);
    cvt(w1p,   fe_w1,  1024, 96,  1024, 80);
    cvt(w2b,   fe_w2,  2560, 1024, 2560, 1024);
    cvt(wih0f, wih[0], 1024, 2560, 1024, 2560);
    cvt(wih0b, wih[1], 1024, 2560, 1024, 2560);
    cvt(wih1f, wih[2], 1024, 512, 1024, 512);
    cvt(wih1b, wih[3], 1024, 512, 1024, 512);
    for (int i = 0; i < 4; ++i) {
        cvt(whhc[i], whh[i], 1024, 256, 1024, 256);
        add_bias_pair<<<4, 256, 0, stream>>>(bcomb[i], bih[i], bhh[i], 1024);
    }
    cvt(fcwp, fc_w, 32, 512, 29, 512);

    // --- feature extractor ---
    gemm_bf16_wmma<true, true><<<dim3(4, M / 32), 256, 0, stream>>>(
        xpad, 96, w1p, 96, fe_b1, Y1, 1024, M, 1024, 96, 1024);
    gemm_bf16_wmma<true, true><<<dim3(10, M / 32), 256, 0, stream>>>(
        Y1, 1024, w2b, 1024, fe_b2, Y2, 2560, M, 2560, 1024, 2560);

    // --- layer 0: x-gate GEMMs then recurrence (both directions) ---
    gemm_bf16_wmma<false, false><<<dim3(4, M / 32), 256, 0, stream>>>(
        Y2, 2560, wih0f, 2560, bcomb[0], xgf, 1024, M, 1024, 2560, 1024);
    gemm_bf16_wmma<false, false><<<dim3(4, M / 32), 256, 0, stream>>>(
        Y2, 2560, wih0b, 2560, bcomb[1], xgb, 1024, M, 1024, 2560, 1024);
    lstm_recurrence<<<2, 512, 0, stream>>>(xgf, xgb, whhc[0], whhc[1], concat0, T);

    // --- layer 1 ---
    gemm_bf16_wmma<false, false><<<dim3(4, M / 32), 256, 0, stream>>>(
        concat0, 512, wih1f, 512, bcomb[2], xgf, 1024, M, 1024, 512, 1024);
    gemm_bf16_wmma<false, false><<<dim3(4, M / 32), 256, 0, stream>>>(
        concat0, 512, wih1b, 512, bcomb[3], xgb, 1024, M, 1024, 512, 1024);
    lstm_recurrence<<<2, 512, 0, stream>>>(xgf, xgb, whhc[2], whhc[3], concat1, T);

    // --- final FC: [M,512] x [29,512]^T + b ---
    gemm_bf16_wmma<false, false><<<dim3(1, M / 32), 256, 0, stream>>>(
        concat1, 512, fcwp, 512, fc_b, (float*)d_out, 29, M, 32, 512, 29);
}
